// MultiHeadSelfAttentionXL_37452114821427
// MI455X (gfx1250) — matrix operand order repaired and checked by reading, not compiled, not run on hardware
//
#include <hip/hip_runtime.h>
#include <hip/hip_bf16.h>

#define DEV __device__ __forceinline__

constexpr int B_  = 2;
constexpr int N_  = 2048;
constexpr int H_  = 2048;
constexpr int HPN = 4096;   // H + N
constexpr int E_  = 512;
constexpr int NH_ = 8;
constexpr int D_  = 64;
constexpr int R_  = HPN + N_;  // 6144 rel positions

typedef __bf16 bf16;
typedef __attribute__((ext_vector_type(16))) __bf16 v16bf;
typedef __attribute__((ext_vector_type(8)))  __bf16 v8bf;
typedef __attribute__((ext_vector_type(8)))  float  v8f;

union V16U { v16bf v; v8bf h[2]; };

DEV v8f vzero8() {
    v8f z;
#pragma unroll
    for (int i = 0; i < 8; ++i) z[i] = 0.0f;
    return z;
}

// A-fragment (16x32 bf16, M=lane%16): one contiguous 16-element run per lane (k = 16*half + s).
DEV v16bf load_afrag(const bf16* p /* &row[k0] */, int half) {
    V16U u;
    u.h[0] = *(const v8bf*)(p + half * 16);
    u.h[1] = *(const v8bf*)(p + half * 16 + 8);
    return u.v;
}

DEV v8f wmma_bf16(v16bf a, v16bf b, v8f c) {
    return __builtin_amdgcn_wmma_f32_16x16x32_bf16(
        false, a, false, b, (short)0, c, false, false);
}

// Async global->LDS copy (ASYNCcnt path, gfx1250). 16B per lane.
DEV void async_g2l_b128(const void* lds_ptr, const void* gaddr) {
    unsigned lds_addr = (unsigned)(unsigned long)lds_ptr;  // low 32b = LDS offset
    asm volatile("global_load_async_to_lds_b128 %0, %1, off"
                 :: "v"(lds_addr), "v"(gaddr)
                 : "memory");
}
DEV void wait_async_le2() {
    asm volatile("s_wait_asynccnt 2" ::: "memory");
}

// Shared double-buffered 16xK(512) x Kx64 GEMM tile: acc[4] over 4 col-groups.
DEV void gemm_tile512(const bf16* arow, const bf16* w0, const bf16* w1,
                      const bf16* w2, const bf16* w3, int half, v8f acc[4]) {
    v16bf a  = load_afrag(arow, half);
    v16bf b0 = *(const v16bf*)(w0);
    v16bf b1 = *(const v16bf*)(w1);
    v16bf b2 = *(const v16bf*)(w2);
    v16bf b3 = *(const v16bf*)(w3);
    for (int kc = 0; kc < 512; kc += 32) {
        int kn = (kc + 32 < 512) ? kc + 32 : kc;  // harmless reload on last iter
        v16bf an = load_afrag(arow + kn, half);
        v16bf n0 = *(const v16bf*)(w0 + kn);
        v16bf n1 = *(const v16bf*)(w1 + kn);
        v16bf n2 = *(const v16bf*)(w2 + kn);
        v16bf n3 = *(const v16bf*)(w3 + kn);
        acc[0] = wmma_bf16(a, b0, acc[0]);
        acc[1] = wmma_bf16(a, b1, acc[1]);
        acc[2] = wmma_bf16(a, b2, acc[2]);
        acc[3] = wmma_bf16(a, b3, acc[3]);
        a = an; b0 = n0; b1 = n1; b2 = n2; b3 = n3;
    }
}

// ---------------- prep kernels ----------------

__global__ void prep_allx_kernel(const float* __restrict__ x,
                                 const float* __restrict__ hist,
                                 bf16* __restrict__ allx) {
    long total = (long)B_ * HPN * E_;
    for (long i = blockIdx.x * (long)blockDim.x + threadIdx.x; i < total;
         i += (long)gridDim.x * blockDim.x) {
        int e = (int)(i % E_);
        long t = i / E_;
        int r = (int)(t % HPN);
        int b = (int)(t / HPN);
        float v = (r < H_) ? hist[((long)b * H_ + r) * E_ + e]
                           : x[((long)b * N_ + (r - H_)) * E_ + e];
        allx[i] = (bf16)v;
    }
}

__global__ void prep_rel_kernel(bf16* __restrict__ rel) {
    long total = (long)R_ * E_;
    for (long i = blockIdx.x * (long)blockDim.x + threadIdx.x; i < total;
         i += (long)gridDim.x * blockDim.x) {
        int e = (int)(i % E_);
        int l = (int)(i / E_);
        int k = (e < E_ / 2) ? e : e - E_ / 2;
        float inv = __powf(10000.0f, -(float)k / (float)(E_ / 2));
        float f = (float)(l - HPN + 1) * inv;
        rel[i] = (bf16)((e < E_ / 2) ? __sinf(f) : __cosf(f));
    }
}

// (NH,E,D) f32 -> (NH,D,E) bf16
__global__ void prep_wt_kernel(const float* __restrict__ w, bf16* __restrict__ wt) {
    long total = (long)NH_ * E_ * D_;
    for (long i = blockIdx.x * (long)blockDim.x + threadIdx.x; i < total;
         i += (long)gridDim.x * blockDim.x) {
        int d = (int)(i % D_);
        int j = (int)((i / D_) % E_);
        int n = (int)(i / ((long)E_ * D_));
        wt[((long)n * D_ + d) * E_ + j] = (bf16)w[i];
    }
}

// w_o (NH,D,E) f32 -> wo_t (E, NH*D) bf16
__global__ void prep_wo_kernel(const float* __restrict__ w, bf16* __restrict__ wot) {
    long total = (long)NH_ * D_ * E_;
    for (long i = blockIdx.x * (long)blockDim.x + threadIdx.x; i < total;
         i += (long)gridDim.x * blockDim.x) {
        int e = (int)(i % E_);
        int d = (int)((i / E_) % D_);
        int n = (int)(i / ((long)D_ * E_));
        wot[(long)e * (NH_ * D_) + n * D_ + d] = (bf16)w[i];
    }
}

// ---------------- projection kernels ----------------

// Q projection + u/v bias + 1/sqrt(64) scale -> qu, qv (bf16, [b][n][row][d])
__global__ __launch_bounds__(256) void proj_q_kernel(
    const bf16* __restrict__ allx, const bf16* __restrict__ wqt,
    const float* __restrict__ ub, const float* __restrict__ vb,
    bf16* __restrict__ qu, bf16* __restrict__ qv) {
    int lane = threadIdx.x & 31, half = lane >> 4, l16 = lane & 15;
    int wid = (int)((blockIdx.x * blockDim.x + threadIdx.x) >> 5);
    if (wid >= B_ * NH_ * (N_ / 16)) return;
    int b = wid >> 10, n = (wid >> 7) & 7, i0 = (wid & 127) << 4;

    const bf16* arow = allx + ((long)b * HPN + H_ + i0 + l16) * E_;
    const bf16* wbase = wqt + (long)n * D_ * E_ + half * 16;
    v8f acc[4];
#pragma unroll
    for (int cg = 0; cg < 4; ++cg) acc[cg] = vzero8();
    gemm_tile512(arow, wbase + (long)(0 + l16) * E_, wbase + (long)(16 + l16) * E_,
                 wbase + (long)(32 + l16) * E_, wbase + (long)(48 + l16) * E_,
                 half, acc);
#pragma unroll
    for (int cg = 0; cg < 4; ++cg) {
        int col = cg * 16 + l16;
        float u = ub[n * D_ + col], vv = vb[n * D_ + col];
#pragma unroll
        for (int v = 0; v < 8; ++v) {
            int row = i0 + v + half * 8;
            long o = ((long)(b * NH_ + n) * N_ + row) * D_ + col;
            qu[o] = (bf16)((acc[cg][v] + u) * 0.125f);
            qv[o] = (bf16)((acc[cg][v] + vv) * 0.125f);
        }
    }
}

// K projection (row-major out)
__global__ __launch_bounds__(256) void proj_k_kernel(
    const bf16* __restrict__ allx, const bf16* __restrict__ wkt,
    bf16* __restrict__ kout) {
    int lane = threadIdx.x & 31, half = lane >> 4, l16 = lane & 15;
    int wid = (int)((blockIdx.x * blockDim.x + threadIdx.x) >> 5);
    if (wid >= B_ * NH_ * (HPN / 16)) return;
    int b = wid >> 11, n = (wid >> 8) & 7, r0 = (wid & 255) << 4;

    const bf16* arow = allx + ((long)b * HPN + r0 + l16) * E_;
    const bf16* wbase = wkt + (long)n * D_ * E_ + half * 16;
    v8f acc[4];
#pragma unroll
    for (int cg = 0; cg < 4; ++cg) acc[cg] = vzero8();
    gemm_tile512(arow, wbase + (long)(0 + l16) * E_, wbase + (long)(16 + l16) * E_,
                 wbase + (long)(32 + l16) * E_, wbase + (long)(48 + l16) * E_,
                 half, acc);
#pragma unroll
    for (int cg = 0; cg < 4; ++cg) {
        int col = cg * 16 + l16;
#pragma unroll
        for (int v = 0; v < 8; ++v) {
            int row = r0 + v + half * 8;
            kout[((long)(b * NH_ + n) * HPN + row) * D_ + col] = (bf16)acc[cg][v];
        }
    }
}

// V projection (transposed out: [b][n][d][key])
__global__ __launch_bounds__(256) void proj_v_kernel(
    const bf16* __restrict__ allx, const bf16* __restrict__ wvt,
    bf16* __restrict__ vtout) {
    int lane = threadIdx.x & 31, half = lane >> 4, l16 = lane & 15;
    int wid = (int)((blockIdx.x * blockDim.x + threadIdx.x) >> 5);
    if (wid >= B_ * NH_ * (HPN / 16)) return;
    int b = wid >> 11, n = (wid >> 8) & 7, r0 = (wid & 255) << 4;

    const bf16* arow = allx + ((long)b * HPN + r0 + l16) * E_;
    const bf16* wbase = wvt + (long)n * D_ * E_ + half * 16;
    v8f acc[4];
#pragma unroll
    for (int cg = 0; cg < 4; ++cg) acc[cg] = vzero8();
    gemm_tile512(arow, wbase + (long)(0 + l16) * E_, wbase + (long)(16 + l16) * E_,
                 wbase + (long)(32 + l16) * E_, wbase + (long)(48 + l16) * E_,
                 half, acc);
#pragma unroll
    for (int cg = 0; cg < 4; ++cg) {
        int col = cg * 16 + l16;
#pragma unroll
        for (int v = 0; v < 8; ++v) {
            int row = r0 + v + half * 8;
            vtout[((long)(b * NH_ + n) * D_ + col) * HPN + row] = (bf16)acc[cg][v];
        }
    }
}

// BD_k = rel_embed @ w_kr -> bdk [n][r][d] bf16
__global__ __launch_bounds__(256) void proj_bdk_kernel(
    const bf16* __restrict__ rel, const bf16* __restrict__ wkrt,
    bf16* __restrict__ bdk) {
    int lane = threadIdx.x & 31, half = lane >> 4, l16 = lane & 15;
    int wid = (int)((blockIdx.x * blockDim.x + threadIdx.x) >> 5);
    if (wid >= NH_ * (R_ / 16)) return;
    int n = wid / (R_ / 16), r0 = (wid % (R_ / 16)) << 4;

    const bf16* arow = rel + (long)(r0 + l16) * E_;
    const bf16* wbase = wkrt + (long)n * D_ * E_ + half * 16;
    v8f acc[4];
#pragma unroll
    for (int cg = 0; cg < 4; ++cg) acc[cg] = vzero8();
    gemm_tile512(arow, wbase + (long)(0 + l16) * E_, wbase + (long)(16 + l16) * E_,
                 wbase + (long)(32 + l16) * E_, wbase + (long)(48 + l16) * E_,
                 half, acc);
#pragma unroll
    for (int cg = 0; cg < 4; ++cg) {
        int col = cg * 16 + l16;
#pragma unroll
        for (int v = 0; v < 8; ++v) {
            int row = r0 + v + half * 8;
            bdk[((long)n * R_ + row) * D_ + col] = (bf16)acc[cg][v];
        }
    }
}

// ---------------- flash attention with rel-shift ----------------
// 8 waves/block share one (b, head); K/V tiles staged via async global->LDS
// with double buffering (ASYNCcnt), shared by all 8 waves.

__global__ __launch_bounds__(256) void attn_kernel(
    const bf16* __restrict__ qu, const bf16* __restrict__ qv,
    const bf16* __restrict__ kbf, const bf16* __restrict__ vtbf,
    const bf16* __restrict__ bdk, const float* __restrict__ mask,
    bf16* __restrict__ ctx) {
    __shared__ __align__(64) float Mbuf[8][16][48];
    __shared__ __align__(64) bf16  Pbuf[8][16][32];
    __shared__ __align__(64) bf16  Kbuf[2][32][64];   // [buf][key][d]
    __shared__ __align__(64) bf16  Vbuf[2][64][32];   // [buf][d][key]

    int tid = threadIdx.x;
    int lane = tid & 31, half = lane >> 4, l16 = lane & 15;
    int ws = tid >> 5;
    int wid = blockIdx.x * 8 + ws;
    int b = wid >> 10, n = (wid >> 7) & 7, i0 = (wid & 127) << 4;

    const bf16* kb = kbf + (long)(b * NH_ + n) * HPN * D_;
    const bf16* vt = vtbf + (long)(b * NH_ + n) * D_ * HPN;
    const bf16* bd = bdk + (long)n * R_ * D_;
    const float* mk = mask + (long)b * N_ * HPN;

    // cooperative staging slices (16B per thread per tile per matrix)
    int krow = tid >> 3, kseg = (tid & 7) * 8;
    int vrow = tid >> 2, vseg = (tid & 3) * 8;

    // prologue: stage tile 0 into buffer 0
    async_g2l_b128(&Kbuf[0][krow][kseg], kb + (long)krow * D_ + kseg);
    async_g2l_b128(&Vbuf[0][vrow][vseg], vt + (long)vrow * HPN + vseg);

    const bf16* qurow = qu + ((long)(b * NH_ + n) * N_ + i0 + l16) * D_;
    const bf16* qvrow = qv + ((long)(b * NH_ + n) * N_ + i0 + l16) * D_;
    v16bf aqu[2], aqv[2];
    aqu[0] = load_afrag(qurow, half);      aqu[1] = load_afrag(qurow + 32, half);
    aqv[0] = load_afrag(qvrow, half);      aqv[1] = load_afrag(qvrow + 32, half);

    float mrun[8], lrun[8];
#pragma unroll
    for (int v = 0; v < 8; ++v) { mrun[v] = -3.0e38f; lrun[v] = 0.0f; }
    v8f O[4];
#pragma unroll
    for (int dc = 0; dc < 4; ++dc) O[dc] = vzero8();

    for (int kt = 0; kt < HPN; kt += 32) {
        int cur = (kt >> 5) & 1;
        int ktn = (kt + 32 < HPN) ? kt + 32 : kt;  // clamped (dummy on last)
        // issue next tile into other buffer
        async_g2l_b128(&Kbuf[cur ^ 1][krow][kseg],
                       kb + (long)(ktn + krow) * D_ + kseg);
        async_g2l_b128(&Vbuf[cur ^ 1][vrow][vseg],
                       vt + (long)vrow * HPN + ktn + vseg);
        wait_async_le2();      // current tile's 2 async ops complete
        __syncthreads();       // visible to all waves

        // AC scores: 16 q-rows x 32 keys from LDS K tile
        v8f s0 = vzero8(), s1 = vzero8();
        {
            const bf16* kr0 = &Kbuf[cur][l16][half * 16];
            const bf16* kr1 = &Kbuf[cur][16 + l16][half * 16];
#pragma unroll
            for (int kc = 0; kc < 2; ++kc) {
                s0 = wmma_bf16(aqu[kc], *(const v16bf*)(kr0 + kc * 32), s0);
                s1 = wmma_bf16(aqu[kc], *(const v16bf*)(kr1 + kc * 32), s1);
            }
        }

        // BD band: M[di][t] = (q+v).BDk[rbase+t], t in [0,48)
        int rbase = kt + (N_ - 1) - i0 - 15;
        {
            int rp = rbase + 32 + l16;                     // prefetch next band
            rp = rp < 0 ? 0 : (rp > R_ - 1 ? R_ - 1 : rp);
            __builtin_prefetch(bd + (long)rp * D_, 0, 1);
        }
#pragma unroll
        for (int t0i = 0; t0i < 3; ++t0i) {
            int r = rbase + t0i * 16 + l16;
            r = r < 0 ? 0 : (r > R_ - 1 ? R_ - 1 : r);
            const bf16* br = bd + (long)r * D_ + half * 16;
            v8f m = vzero8();
            m = wmma_bf16(aqv[0], *(const v16bf*)br, m);
            m = wmma_bf16(aqv[1], *(const v16bf*)(br + 32), m);
#pragma unroll
            for (int v = 0; v < 8; ++v)
                Mbuf[ws][v + half * 8][t0i * 16 + l16] = m[v];
        }
        __syncthreads();

        // rel-shift gather + mask + streaming softmax
#pragma unroll
        for (int v = 0; v < 8; ++v) {
            int di = v + half * 8;
            const float* mr = mk + (long)(i0 + di) * HPN + kt;
            float a0 = s0[v] + Mbuf[ws][di][l16 - di + 15];
            float a1 = s1[v] + Mbuf[ws][di][l16 + 31 - di];
            float m0 = mr[l16], m1 = mr[16 + l16];
            a0 = a0 * m0 - 1.0e8f * (1.0f - m0);
            a1 = a1 * m1 - 1.0e8f * (1.0f - m1);
            float mx = fmaxf(a0, a1);
            mx = fmaxf(mx, __shfl_xor(mx, 1));
            mx = fmaxf(mx, __shfl_xor(mx, 2));
            mx = fmaxf(mx, __shfl_xor(mx, 4));
            mx = fmaxf(mx, __shfl_xor(mx, 8));
            float mnew = fmaxf(mrun[v], mx);
            float p0 = __expf(a0 - mnew), p1 = __expf(a1 - mnew);
            float rs = p0 + p1;
            rs += __shfl_xor(rs, 1);
            rs += __shfl_xor(rs, 2);
            rs += __shfl_xor(rs, 4);
            rs += __shfl_xor(rs, 8);
            float sc = __expf(mrun[v] - mnew);
            lrun[v] = lrun[v] * sc + rs;
            mrun[v] = mnew;
#pragma unroll
            for (int dc = 0; dc < 4; ++dc) O[dc][v] *= sc;
            Pbuf[ws][di][l16] = (bf16)p0;
            Pbuf[ws][di][16 + l16] = (bf16)p1;
        }
        __syncthreads();

        // O += P @ V  (P through LDS into A-fragment, V from LDS tile)
        v16bf pf = load_afrag(&Pbuf[ws][l16][0], half);
#pragma unroll
        for (int dc = 0; dc < 4; ++dc) {
            const bf16* vr = &Vbuf[cur][dc * 16 + l16][half * 16];
            O[dc] = wmma_bf16(pf, *(const v16bf*)vr, O[dc]);
        }
        __syncthreads();
    }

    // ctx [b][row][n*64+d] bf16
#pragma unroll
    for (int dc = 0; dc < 4; ++dc) {
        int col = dc * 16 + l16;
#pragma unroll
        for (int v = 0; v < 8; ++v) {
            int row = i0 + v + half * 8;
            ctx[((long)b * N_ + row) * (NH_ * D_) + n * D_ + col] =
                (bf16)(O[dc][v] / lrun[v]);
        }
    }
}

// ---------------- output projection: out = ctx @ w_o_flat ----------------

__global__ __launch_bounds__(256) void outproj_kernel(
    const bf16* __restrict__ ctx, const bf16* __restrict__ wot,
    float* __restrict__ out) {
    int lane = threadIdx.x & 31, half = lane >> 4, l16 = lane & 15;
    int wid = (int)((blockIdx.x * blockDim.x + threadIdx.x) >> 5);
    if (wid >= (B_ * N_ / 16) * (E_ / 64)) return;
    int rt = wid >> 3, ct = wid & 7;

    const bf16* arow = ctx + (long)(rt * 16 + l16) * (NH_ * D_);
    const bf16* wbase = wot + (long)(ct * 64) * (NH_ * D_) + half * 16;
    v8f acc[4];
#pragma unroll
    for (int cg = 0; cg < 4; ++cg) acc[cg] = vzero8();
    gemm_tile512(arow, wbase + (long)(0 + l16) * (NH_ * D_),
                 wbase + (long)(16 + l16) * (NH_ * D_),
                 wbase + (long)(32 + l16) * (NH_ * D_),
                 wbase + (long)(48 + l16) * (NH_ * D_), half, acc);
#pragma unroll
    for (int cg = 0; cg < 4; ++cg) {
#pragma unroll
        for (int v = 0; v < 8; ++v) {
            int row = rt * 16 + v + half * 8;
            out[(long)row * E_ + ct * 64 + cg * 16 + l16] = acc[cg][v];
        }
    }
}

// ---------------- launcher ----------------

extern "C" void kernel_launch(void* const* d_in, const int* in_sizes, int n_in,
                              void* d_out, int out_size, void* d_ws, size_t ws_size,
                              hipStream_t stream) {
    (void)in_sizes; (void)n_in; (void)out_size; (void)ws_size;

    const float* x    = (const float*)d_in[0];
    const float* hist = (const float*)d_in[1];
    const float* mask = (const float*)d_in[2];
    const float* w_q  = (const float*)d_in[3];
    const float* w_k  = (const float*)d_in[4];
    const float* w_v  = (const float*)d_in[5];
    const float* w_kr = (const float*)d_in[6];
    const float* w_o  = (const float*)d_in[7];
    const float* ub   = (const float*)d_in[8];
    const float* vb   = (const float*)d_in[9];
    float* out = (float*)d_out;

    char* wsp = (char*)d_ws;
    auto carve = [&](size_t bytes) {
        void* p = (void*)wsp;
        wsp += (bytes + 255) & ~(size_t)255;
        return p;
    };
    bf16* allx = (bf16*)carve((size_t)B_ * HPN * E_ * 2);
    bf16* rel  = (bf16*)carve((size_t)R_ * E_ * 2);
    bf16* wqt  = (bf16*)carve((size_t)NH_ * D_ * E_ * 2);
    bf16* wkt  = (bf16*)carve((size_t)NH_ * D_ * E_ * 2);
    bf16* wvt  = (bf16*)carve((size_t)NH_ * D_ * E_ * 2);
    bf16* wkrt = (bf16*)carve((size_t)NH_ * D_ * E_ * 2);
    bf16* wot  = (bf16*)carve((size_t)E_ * NH_ * D_ * 2);
    bf16* qu   = (bf16*)carve((size_t)B_ * NH_ * N_ * D_ * 2);
    bf16* qv   = (bf16*)carve((size_t)B_ * NH_ * N_ * D_ * 2);
    bf16* kbf  = (bf16*)carve((size_t)B_ * NH_ * HPN * D_ * 2);
    bf16* vtbf = (bf16*)carve((size_t)B_ * NH_ * D_ * HPN * 2);
    bf16* bdk  = (bf16*)carve((size_t)NH_ * R_ * D_ * 2);
    bf16* ctx  = (bf16*)carve((size_t)B_ * N_ * NH_ * D_ * 2);

    prep_allx_kernel<<<4096, 256, 0, stream>>>(x, hist, allx);
    prep_rel_kernel<<<4096, 256, 0, stream>>>(rel);
    prep_wt_kernel<<<1024, 256, 0, stream>>>(w_q, wqt);
    prep_wt_kernel<<<1024, 256, 0, stream>>>(w_k, wkt);
    prep_wt_kernel<<<1024, 256, 0, stream>>>(w_v, wvt);
    prep_wt_kernel<<<1024, 256, 0, stream>>>(w_kr, wkrt);
    prep_wo_kernel<<<1024, 256, 0, stream>>>(w_o, wot);

    proj_q_kernel<<<256, 256, 0, stream>>>(allx, wqt, ub, vb, qu, qv);
    proj_k_kernel<<<512, 256, 0, stream>>>(allx, wkt, kbf);
    proj_v_kernel<<<512, 256, 0, stream>>>(allx, wvt, vtbf);
    proj_bdk_kernel<<<384, 256, 0, stream>>>(rel, wkrt, bdk);

    attn_kernel<<<256, 256, 0, stream>>>(qu, qv, kbf, vtbf, bdk, mask, ctx);

    outproj_kernel<<<256, 256, 0, stream>>>(ctx, wot, out);
}
